// Attention_24996709663409
// MI455X (gfx1250) — matrix operand order repaired and checked by reading, not compiled, not run on hardware
//
#include <hip/hip_runtime.h>
#include <hip/hip_fp16.h>
#include <stdint.h>

typedef __attribute__((ext_vector_type(16))) _Float16 v16h;
typedef __attribute__((ext_vector_type(8)))  float    v8f;
typedef __attribute__((ext_vector_type(4)))  int      i4;

#define BB     2
#define NN     32768
#define DIMD   128
#define HEADS  4
#define DHH    32
#define WW     128
#define NWW    256
#define MM     4
#define KPAD   288      /* 4 mem + 256 window + 28 pad -> multiple of 32 */
#define QSCALE 0.17677669529663687f  /* 32^-0.5 */

union AF { v16h h; uint32_t u[8]; };

typedef __attribute__((address_space(1))) i4 gas_i4;
typedef __attribute__((address_space(3))) i4 las_i4;

// A-fragment (16x32 f16) K index for VGPR v, lane-half 'half' (contiguous: k = 16*half + 2*v)
__device__ __forceinline__ int a_kidx(int v, int half) {
  return half * 16 + 2 * v;
}

// ---------------------------------------------------------------------------
// Kernel 1: fused projections  P = seq @ [Wq | Wk | Wv | Wg]
// q gets (+bq)*DH^-0.5 folded in; all outputs stored f16.
// block = 128 thr (4 waves), tile = 64 rows x 64 cols, K = 128.
// ---------------------------------------------------------------------------
__global__ __launch_bounds__(128) void proj_kernel(
    const float* __restrict__ seq, const float* __restrict__ Wq,
    const float* __restrict__ bq,  const float* __restrict__ Wkv,
    const float* __restrict__ Wg,
    _Float16* __restrict__ qf, _Float16* __restrict__ kf,
    _Float16* __restrict__ vf, _Float16* __restrict__ gf)
{
  __shared__ _Float16 Xl[64 * 128];   // [row][k]
  __shared__ _Float16 Wt[64 * 128];   // [col][k]  (transposed for B frags)

  const int t = threadIdx.x;
  const int rowbase = blockIdx.x * 64;
  const int colbase = blockIdx.y * 64;

  for (int idx = t; idx < 64 * 128; idx += 128) {
    int r = idx >> 7, k = idx & 127;
    Xl[idx] = (_Float16)seq[(size_t)(rowbase + r) * DIMD + k];
  }
  for (int idx = t; idx < 64 * 128; idx += 128) {
    int c = idx >> 7, k = idx & 127;
    int cg = colbase + c;
    float wv;
    if (cg < 128)      wv = Wq[k * 128 + cg];
    else if (cg < 384) wv = Wkv[k * 256 + (cg - 128)];
    else               wv = Wg[k * 128 + (cg - 384)];
    Wt[idx] = (_Float16)wv;
  }
  __syncthreads();

  const int w = t >> 5, lane = t & 31, half = lane >> 4, ln = lane & 15;

  v8f acc[4] = {};
  for (int kb = 0; kb < 128; kb += 32) {
    AF a;
#pragma unroll
    for (int v = 0; v < 8; ++v)
      a.u[v] = *(const uint32_t*)&Xl[(w * 16 + ln) * 128 + kb + a_kidx(v, half)];
#pragma unroll
    for (int ct = 0; ct < 4; ++ct) {
      AF bf;
#pragma unroll
      for (int v = 0; v < 8; ++v)
        bf.u[v] = *(const uint32_t*)&Wt[(ct * 16 + ln) * 128 + kb + half * 16 + 2 * v];
      acc[ct] = __builtin_amdgcn_wmma_f32_16x16x32_f16(
          false, a.h, false, bf.h, (short)0, acc[ct], false, false);
    }
  }

#pragma unroll
  for (int ct = 0; ct < 4; ++ct) {
#pragma unroll
    for (int i = 0; i < 8; ++i) {
      int r  = rowbase + w * 16 + i + 8 * half;
      int cg = colbase + ct * 16 + ln;
      int b = r >> 15, n = r & (NN - 1);
      float val = acc[ct][i];
      if (cg < 128) {
        int hh = cg >> 5, d = cg & 31;
        val = (val + bq[cg]) * QSCALE;
        qf[((((size_t)b * HEADS + hh) * NN + n) << 5) + d] = (_Float16)val;
      } else if (cg < 256) {
        int c2 = cg - 128, hh = c2 >> 5, d = c2 & 31;
        kf[((((size_t)b * HEADS + hh) * NN + n) << 5) + d] = (_Float16)val;
      } else if (cg < 384) {
        int c2 = cg - 256, hh = c2 >> 5, d = c2 & 31;
        vf[((((size_t)b * HEADS + hh) * NN + n) << 5) + d] = (_Float16)val;
      } else {
        gf[((size_t)r << 7) + (cg - 384)] = (_Float16)val;
      }
    }
  }
}

// ---------------------------------------------------------------------------
// Kernel 2: windowed flash attention. One block per (b, h, nw).
// Keys = [4 memory | prev window 128 | cur window 128 | pad] -> 288.
// Window K/V rows staged via GLOBAL_LOAD_ASYNC_TO_LDS_B128 (ASYNCcnt),
// overlapped with mask/memory-kv/pad scalar work and q-fragment loads.
// ---------------------------------------------------------------------------
__global__ __launch_bounds__(128) void attn_kernel(
    const _Float16* __restrict__ qf, const _Float16* __restrict__ kf,
    const _Float16* __restrict__ vf, const float* __restrict__ bias,
    const unsigned char* __restrict__ mask, const float* __restrict__ mem_kv,
    _Float16* __restrict__ AO)
{
  __shared__ _Float16 Kl[KPAD * 32];    // [j][dh]
  __shared__ _Float16 Vr[2 * WW * 32];  // async staging, row-major [win j][dh]
  __shared__ _Float16 Vt[32 * KPAD];    // [dh][j]  (transposed)
  __shared__ float    madd[KPAD];       // 0 or -1e38 mask add
  __shared__ _Float16 Pl[4 * 16 * 32];  // per-wave P re-layout staging

  const int blk = blockIdx.x;
  const int b  = blk / (HEADS * NWW);
  const int h  = (blk / NWW) % HEADS;
  const int nw = blk % NWW;
  const int t  = threadIdx.x;

  // --- Phase 1: issue async copies of the 256 window K/V rows (16B chunks).
  // nw==0 previous-window rows are address-clamped; their columns are nulled
  // by madd, and clamped reads return finite f16 (no NaN poisoning).
  for (int idx = t; idx < 2 * WW * 4; idx += 128) {
    int off = idx >> 2;     // window row 0..255
    int ch  = idx & 3;      // 16B chunk within the 64B row
    int tok = (off < WW) ? ((nw - 1) * WW + off) : (nw * WW + (off - WW));
    tok = tok < 0 ? 0 : (tok >= NN ? NN - 1 : tok);
    size_t base = ((((size_t)b * HEADS + h) * NN + tok) << 5) + ch * 8;
    __builtin_amdgcn_global_load_async_to_lds_b128(
        (gas_i4*)(kf + base), (las_i4*)&Kl[(MM + off) * 32 + ch * 8], 0, 0);
    __builtin_amdgcn_global_load_async_to_lds_b128(
        (gas_i4*)(vf + base), (las_i4*)&Vr[off * 32 + ch * 8], 0, 0);
  }

  // --- Phase 2 (overlapped with async): memory-kv rows, pad rows, mask adds.
  for (int jg = t; jg < KPAD; jg += 128) {
    float ma = -1e38f;
    if (jg < MM) {
      ma = 0.f;
      for (int d = 0; d < DHH; ++d) {
        Kl[jg * 32 + d]   = (_Float16)mem_kv[(((size_t)0 * HEADS + h) * MM + jg) * DHH + d];
        Vt[d * KPAD + jg] = (_Float16)mem_kv[(((size_t)1 * HEADS + h) * MM + jg) * DHH + d];
      }
    } else if (jg < MM + 2 * WW) {
      int off = jg - MM;
      if (off < WW) ma = ((nw > 0) && (mask[(size_t)b * NN + (nw - 1) * WW + off] != 0)) ? 0.f : -1e38f;
      else          ma = (mask[(size_t)b * NN + nw * WW + (off - WW)] != 0) ? 0.f : -1e38f;
    } else {
      // pad rows/cols must be finite (uninitialized LDS could hold NaN bits)
      for (int d = 0; d < DHH; ++d) { Kl[jg * 32 + d] = (_Float16)0.f; Vt[d * KPAD + jg] = (_Float16)0.f; }
    }
    madd[jg] = ma;
  }

  const int w = t >> 5, lane = t & 31, half = lane >> 4, ln = lane & 15;

  // q A-fragments (overlapped too): wave w owns query row-tiles 2w and 2w+1
  AF aq[2];
#pragma unroll
  for (int qt = 0; qt < 2; ++qt) {
    int qrow = nw * WW + (2 * w + qt) * 16 + ln;
    size_t base = (((size_t)b * HEADS + h) * NN + qrow) << 5;
#pragma unroll
    for (int v = 0; v < 8; ++v)
      aq[qt].u[v] = *(const uint32_t*)&qf[base + a_kidx(v, half)];
  }

  asm volatile("s_wait_asynccnt 0" ::: "memory");
  __syncthreads();

  // --- Phase 3: LDS->LDS transpose of V window rows into Vt.
  for (int off = t; off < 2 * WW; off += 128)
    for (int d = 0; d < DHH; ++d)
      Vt[d * KPAD + MM + off] = Vr[off * 32 + d];
  __syncthreads();

  float m_st[2][8], l_st[2][8];
  v8f o_acc[2][2] = {};
#pragma unroll
  for (int qt = 0; qt < 2; ++qt)
#pragma unroll
    for (int i = 0; i < 8; ++i) { m_st[qt][i] = -1e30f; l_st[qt][i] = 0.f; }

  for (int jb = 0; jb < KPAD; jb += 32) {
    // B fragments of V (shared by both q tiles): B[K=32 keys][N=16 dh]
    AF bv[2];
#pragma unroll
    for (int dt = 0; dt < 2; ++dt)
#pragma unroll
      for (int v = 0; v < 8; ++v)
        bv[dt].u[v] = *(const uint32_t*)&Vt[(dt * 16 + ln) * KPAD + jb + half * 16 + 2 * v];

    for (int qt = 0; qt < 2; ++qt) {
      int qi_base = (2 * w + qt) * 16;
      float sv[2][8], tmax[8];
#pragma unroll
      for (int i = 0; i < 8; ++i) tmax[i] = -1e38f;

#pragma unroll
      for (int sub = 0; sub < 2; ++sub) {
        AF bk;   // B[K=32 dh][N=16 keys]
#pragma unroll
        for (int v = 0; v < 8; ++v)
          bk.u[v] = *(const uint32_t*)&Kl[(jb + sub * 16 + ln) * 32 + half * 16 + 2 * v];
        v8f s = {};
        s = __builtin_amdgcn_wmma_f32_16x16x32_f16(
            false, aq[qt].h, false, bk.h, (short)0, s, false, false);
#pragma unroll
        for (int i = 0; i < 8; ++i) {
          int qi = qi_base + i + 8 * half;
          int jg = jb + sub * 16 + ln;
          float bval = 0.f;
          if (jg >= MM && jg < MM + 2 * WW)
            bval = bias[((((size_t)b * NWW + nw) * WW + qi) << 8) + (jg - MM)];
          float x = s[i] + bval;
          x = 50.f * tanhf(x * 0.02f) + madd[jg];   // softclamp + mask
          sv[sub][i] = x;
          tmax[i] = fmaxf(tmax[i], x);
        }
      }
      // row-max across the 16 lanes holding this row's columns
#pragma unroll
      for (int i = 0; i < 8; ++i) {
        float v = tmax[i];
        v = fmaxf(v, __shfl_xor(v, 1)); v = fmaxf(v, __shfl_xor(v, 2));
        v = fmaxf(v, __shfl_xor(v, 4)); v = fmaxf(v, __shfl_xor(v, 8));
        tmax[i] = v;
      }
      // online softmax update
#pragma unroll
      for (int i = 0; i < 8; ++i) {
        float newm  = fmaxf(m_st[qt][i], tmax[i]);
        float alpha = __expf(m_st[qt][i] - newm);
        m_st[qt][i] = newm;
        float p0 = __expf(sv[0][i] - newm);
        float p1 = __expf(sv[1][i] - newm);
        sv[0][i] = p0; sv[1][i] = p1;
        float r = p0 + p1;
        r += __shfl_xor(r, 1); r += __shfl_xor(r, 2);
        r += __shfl_xor(r, 4); r += __shfl_xor(r, 8);
        l_st[qt][i] = l_st[qt][i] * alpha + r;
        o_acc[qt][0][i] *= alpha;
        o_acc[qt][1][i] *= alpha;
      }
      // re-layout P (C-layout) -> A-layout via per-wave LDS bounce
#pragma unroll
      for (int sub = 0; sub < 2; ++sub)
#pragma unroll
        for (int i = 0; i < 8; ++i)
          Pl[w * 512 + (i + 8 * half) * 32 + sub * 16 + ln] = (_Float16)sv[sub][i];
      asm volatile("s_wait_dscnt 0" ::: "memory");
      AF ap;
#pragma unroll
      for (int v = 0; v < 8; ++v)
        ap.u[v] = *(const uint32_t*)&Pl[w * 512 + ln * 32 + a_kidx(v, half)];

      o_acc[qt][0] = __builtin_amdgcn_wmma_f32_16x16x32_f16(
          false, ap.h, false, bv[0].h, (short)0, o_acc[qt][0], false, false);
      o_acc[qt][1] = __builtin_amdgcn_wmma_f32_16x16x32_f16(
          false, ap.h, false, bv[1].h, (short)0, o_acc[qt][1], false, false);
    }
  }

#pragma unroll
  for (int qt = 0; qt < 2; ++qt)
#pragma unroll
    for (int dt = 0; dt < 2; ++dt)
#pragma unroll
      for (int i = 0; i < 8; ++i) {
        int qi = (2 * w + qt) * 16 + i + 8 * half;
        float val = o_acc[qt][dt][i] / l_st[qt][i];
        AO[(((size_t)b * NN + nw * WW + qi) << 7) + h * 32 + dt * 16 + ln] = (_Float16)val;
      }
}

// ---------------------------------------------------------------------------
// Kernel 3: out = (AO * sigmoid(gate)) @ Wo   [65536x128]@[128x128] -> f32
// ---------------------------------------------------------------------------
__global__ __launch_bounds__(128) void out_kernel(
    const _Float16* __restrict__ AO, const _Float16* __restrict__ gf,
    const float* __restrict__ Wo, float* __restrict__ out)
{
  __shared__ _Float16 Xl[64 * 128];    // gated rows
  __shared__ _Float16 WoT[128 * 128];  // [col][k]

  const int t = threadIdx.x;
  const int rowbase = blockIdx.x * 64;

  for (int idx = t; idx < 64 * 128; idx += 128) {
    int r = idx >> 7, c = idx & 127;
    size_t g = ((size_t)(rowbase + r) << 7) + c;
    float a  = (float)AO[g];
    float gg = (float)gf[g];
    Xl[idx] = (_Float16)(a * (1.f / (1.f + __expf(-gg))));
  }
  for (int idx = t; idx < 128 * 128; idx += 128) {
    int c = idx >> 7, k = idx & 127;
    WoT[idx] = (_Float16)Wo[k * 128 + c];
  }
  __syncthreads();

  const int w = t >> 5, lane = t & 31, half = lane >> 4, ln = lane & 15;

  v8f acc[8] = {};
  for (int kb = 0; kb < 128; kb += 32) {
    AF a;
#pragma unroll
    for (int v = 0; v < 8; ++v)
      a.u[v] = *(const uint32_t*)&Xl[(w * 16 + ln) * 128 + kb + a_kidx(v, half)];
#pragma unroll
    for (int ct = 0; ct < 8; ++ct) {
      AF bf;
#pragma unroll
      for (int v = 0; v < 8; ++v)
        bf.u[v] = *(const uint32_t*)&WoT[(ct * 16 + ln) * 128 + kb + half * 16 + 2 * v];
      acc[ct] = __builtin_amdgcn_wmma_f32_16x16x32_f16(
          false, a.h, false, bf.h, (short)0, acc[ct], false, false);
    }
  }
#pragma unroll
  for (int ct = 0; ct < 8; ++ct)
#pragma unroll
    for (int i = 0; i < 8; ++i) {
      int r = rowbase + w * 16 + i + 8 * half;
      out[((size_t)r << 7) + ct * 16 + ln] = acc[ct][i];
    }
}

// ---------------------------------------------------------------------------
extern "C" void kernel_launch(void* const* d_in, const int* in_sizes, int n_in,
                              void* d_out, int out_size, void* d_ws, size_t ws_size,
                              hipStream_t stream) {
  (void)in_sizes; (void)n_in; (void)out_size; (void)ws_size;
  const float*         seq    = (const float*)d_in[0];
  const unsigned char* mask   = (const unsigned char*)d_in[1];
  const float*         abias  = (const float*)d_in[2];
  const float*         Wq     = (const float*)d_in[3];
  const float*         bq     = (const float*)d_in[4];
  const float*         Wkv    = (const float*)d_in[5];
  const float*         Wg     = (const float*)d_in[6];
  const float*         Wo     = (const float*)d_in[7];
  const float*         mem_kv = (const float*)d_in[8];

  char* ws = (char*)d_ws;
  const size_t HALF_QKV = (size_t)BB * HEADS * NN * DHH * sizeof(_Float16); // 16 MB
  _Float16* qf = (_Float16*)(ws);
  _Float16* kf = (_Float16*)(ws + 1 * HALF_QKV);
  _Float16* vf = (_Float16*)(ws + 2 * HALF_QKV);
  _Float16* gf = (_Float16*)(ws + 3 * HALF_QKV);
  _Float16* AO = (_Float16*)(ws + 4 * HALF_QKV);

  // 1) fused projections: 65536 rows x 512 cols
  dim3 g1((BB * NN) / 64, 512 / 64);
  proj_kernel<<<g1, 128, 0, stream>>>(seq, Wq, bq, Wkv, Wg, qf, kf, vf, gf);

  // 2) windowed attention: one block per (b, h, nw)
  attn_kernel<<<BB * HEADS * NWW, 128, 0, stream>>>(qf, kf, vf, abias, mask,
                                                    mem_kv, AO);

  // 3) gating + output GEMM
  out_kernel<<<(BB * NN) / 64, 128, 0, stream>>>(AO, gf, Wo, (float*)d_out);
}
